// PropertyAggregator_52673478918815
// MI455X (gfx1250) — matrix operand order repaired and checked by reading, not compiled, not run on hardware
//
#include <hip/hip_runtime.h>
#include <hip/hip_bf16.h>

// ---------------------------------------------------------------------------
// PropertyAggregator on MI455X (gfx1250).
// GEMMs via v_wmma_f32_16x16x32_bf16 (fp32 accumulate). Weights transposed +
// downconverted to bf16 (N,K) once per launch; activations produced in bf16
// by each producer. Tile staging uses GLOBAL_LOAD_ASYNC_TO_LDS_B128 with LDS
// double-buffering (no VGPR staging -> no spills), falling back to a simple
// synchronous b128 stage if the async builtin is unavailable.
// ---------------------------------------------------------------------------

typedef __attribute__((ext_vector_type(16))) __bf16 v16bf;
typedef __attribute__((ext_vector_type(8)))  float  v8f;
typedef int v4i_vec __attribute__((vector_size(16)));   // matches builtin param

#define LDT 72   // padded LDS row stride (bf16 elems): 144B rows, 16B aligned

#if __has_builtin(__builtin_amdgcn_global_load_async_to_lds_b128)
#define HAS_ASYNC_LDS 1
#else
#define HAS_ASYNC_LDS 0
#endif

#define GLOBAL_AS __attribute__((address_space(1)))
#define LDS_AS    __attribute__((address_space(3)))

__device__ __forceinline__ unsigned short f2bf(float f) {
  union { float f; unsigned u; } c; c.f = f;
  unsigned u = c.u;
  unsigned r = (u + 0x7FFFu + ((u >> 16) & 1u)) >> 16;   // round-nearest-even
  return (unsigned short)r;
}

union FragU { v16bf v; uint4 q[2]; };

#if HAS_ASYNC_LDS
__device__ __forceinline__ void async_cp16(const unsigned short* g, unsigned short* l) {
  __builtin_amdgcn_global_load_async_to_lds_b128(
      (GLOBAL_AS v4i_vec*)g, (LDS_AS v4i_vec*)l, 0, 0);
}
#endif

__device__ __forceinline__ void wait_async0() {
#if __has_builtin(__builtin_amdgcn_s_wait_asynccnt)
  __builtin_amdgcn_s_wait_asynccnt(0);
#else
  asm volatile("s_wait_asynccnt 0" ::: "memory");
#endif
}

// 8 WMMAs x 2 sub-steps on one staged 128x128x64 tile pair.
__device__ __forceinline__ void do_tile(
    const unsigned short* lA, const unsigned short* lB,
    int wm, int wn, int half, int l15, v8f acc[8])
{
#pragma unroll
  for (int s = 0; s < 2; ++s) {
    const int ko = s * 32;
    // B frag: lane = column n, lane-half selects 16 contiguous K
    FragU bfr[2];
#pragma unroll
    for (int nt = 0; nt < 2; ++nt) {
      const uint4* p = (const uint4*)&lB[(wn * 32 + nt * 16 + l15) * LDT + ko + half * 16];
      bfr[nt].q[0] = p[0]; bfr[nt].q[1] = p[1];
    }
#pragma unroll
    for (int mt = 0; mt < 4; ++mt) {
      // A frag: lane = row m, lane-half selects 16 contiguous K
      FragU af;
      const unsigned short* pa = &lA[(wm * 64 + mt * 16 + l15) * LDT + ko + half * 16];
      af.q[0] = *(const uint4*)pa;
      af.q[1] = *(const uint4*)(pa + 8);
#pragma unroll
      for (int nt = 0; nt < 2; ++nt)
        acc[mt * 2 + nt] = __builtin_amdgcn_wmma_f32_16x16x32_bf16(
            false, af.v, false, bfr[nt].v, (short)0, acc[mt * 2 + nt], false, false);
    }
  }
}

// ---------------------------------------------------------------------------
// GEMM: C[m,n] = A[m,:K](bf16) @ Bt[n,:K](bf16, pre-transposed) + bias[n]
// Optional ReLU; optional fp32 and/or bf16 outputs. z-batched for proj.
// ---------------------------------------------------------------------------
__global__ __launch_bounds__(256) void gemm_bf16_v3(
    const unsigned short* __restrict__ A, long lda, long zA,
    const unsigned short* __restrict__ Bt, long zB,
    const float* __restrict__ bias, long zBias,
    float* __restrict__ Cf, long ldcf, long zCf,
    unsigned short* __restrict__ Cb, long ldcb, long zCb,
    int K, int relu)
{
  const int tid  = threadIdx.x;
  const int lane = tid & 31;
  const int wave = tid >> 5;
  const int wm   = wave >> 2;      // 0..1
  const int wn   = wave & 3;       // 0..3
  const int z    = blockIdx.z;
  const int m0   = blockIdx.y * 128;
  const int n0   = blockIdx.x * 128;

  const unsigned short* Ab = A  + (long)z * zA;
  const unsigned short* Bb = Bt + (long)z * zB;

  const int half = lane >> 4;
  const int l15  = lane & 15;

  // staging coords: thread copies 4x 8 contiguous bf16 per tile side
  int rows[4], kcs[4];
#pragma unroll
  for (int i = 0; i < 4; ++i) {
    int e = tid + i * 256;         // 0..1023
    rows[i] = e >> 3;              // 0..127
    kcs[i]  = (e & 7) << 3;        // 0..56
  }

  v8f acc[8] = {};

#if HAS_ASYNC_LDS
  __shared__ unsigned short ldsA[2][128 * LDT];
  __shared__ unsigned short ldsB[2][128 * LDT];

  // issue tile 0
#pragma unroll
  for (int i = 0; i < 4; ++i) {
    async_cp16(Ab + (long)(m0 + rows[i]) * lda + kcs[i], &ldsA[0][rows[i] * LDT + kcs[i]]);
    async_cp16(Bb + (long)(n0 + rows[i]) * K   + kcs[i], &ldsB[0][rows[i] * LDT + kcs[i]]);
  }
  wait_async0();
  __syncthreads();

  const int nT = K >> 6;
  for (int t = 0; t < nT; ++t) {
    const int cur = t & 1, nxt = cur ^ 1;
    if (t + 1 < nT) {              // DMA next tile while WMMAs run
      const long ko = (long)(t + 1) << 6;
#pragma unroll
      for (int i = 0; i < 4; ++i) {
        async_cp16(Ab + (long)(m0 + rows[i]) * lda + ko + kcs[i],
                   &ldsA[nxt][rows[i] * LDT + kcs[i]]);
        async_cp16(Bb + (long)(n0 + rows[i]) * K   + ko + kcs[i],
                   &ldsB[nxt][rows[i] * LDT + kcs[i]]);
      }
    }
    do_tile(ldsA[cur], ldsB[cur], wm, wn, half, l15, acc);
    wait_async0();                 // my next-tile DMAs landed
    __syncthreads();               // everyone done reading cur + loading nxt
  }
#else
  __shared__ unsigned short ldsA1[128 * LDT];
  __shared__ unsigned short ldsB1[128 * LDT];

  for (int k0 = 0; k0 < K; k0 += 64) {
#pragma unroll
    for (int i = 0; i < 4; ++i) {
      uint4 va = *(const uint4*)(Ab + (long)(m0 + rows[i]) * lda + (k0 + kcs[i]));
      uint4 vb = *(const uint4*)(Bb + (long)(n0 + rows[i]) * K   + (k0 + kcs[i]));
      *(uint4*)&ldsA1[rows[i] * LDT + kcs[i]] = va;
      *(uint4*)&ldsB1[rows[i] * LDT + kcs[i]] = vb;
    }
    __syncthreads();
    do_tile(ldsA1, ldsB1, wm, wn, half, l15, acc);
    __syncthreads();
  }
#endif

  const float* biasb = bias + (long)z * zBias;
  float* Cfp = Cf ? Cf + (long)z * zCf : nullptr;
  unsigned short* Cbp = Cb ? Cb + (long)z * zCb : nullptr;
#pragma unroll
  for (int mt = 0; mt < 4; ++mt) {
#pragma unroll
    for (int nt = 0; nt < 2; ++nt) {
      int col = n0 + wn * 32 + nt * 16 + l15;
      float bvv = biasb[col];
#pragma unroll
      for (int i = 0; i < 8; ++i) {
        int row = m0 + wm * 64 + mt * 16 + i + half * 8;
        float v = acc[mt * 2 + nt][i] + bvv;
        if (relu) v = fmaxf(v, 0.0f);
        if (Cfp) Cfp[(long)row * ldcf + col] = v;
        if (Cbp) Cbp[(long)row * ldcb + col] = f2bf(v);
      }
    }
  }
}

// Transpose + downconvert weights: W(K,N) fp32 -> out(N,K) bf16. z-batched.
__global__ __launch_bounds__(256) void wT_kernel(
    const float* __restrict__ W, unsigned short* __restrict__ out, int K, int N)
{
  __shared__ float t[32][33];
  int z = blockIdx.z;
  const float* Wz = W + (long)z * K * N;
  unsigned short* oz = out + (long)z * K * N;
  int kb = blockIdx.y * 32, nb = blockIdx.x * 32;
  int tx = threadIdx.x, ty = threadIdx.y;          // (32, 8)
#pragma unroll
  for (int i = 0; i < 32; i += 8)
    t[ty + i][tx] = Wz[(long)(kb + ty + i) * N + (nb + tx)];
  __syncthreads();
#pragma unroll
  for (int i = 0; i < 32; i += 8)
    oz[(long)(nb + ty + i) * K + (kb + tx)] = f2bf(t[tx][ty + i]);
}

// Flat fp32 -> bf16 (2 elements/thread, packed 32-bit store).
__global__ __launch_bounds__(256) void cvt_kernel(
    const float* __restrict__ in, unsigned short* __restrict__ out)
{
  long i = ((long)blockIdx.x * 256 + threadIdx.x) * 2;
  float2 v = *(const float2*)(in + i);
  unsigned a = f2bf(v.x), b = f2bf(v.y);
  *(unsigned*)(out + i) = a | (b << 16);
}

// Attention over P=3 tokens, HD=96: one wave per (sequence, head). bf16 out.
__global__ __launch_bounds__(256) void attn_kernel(
    const float* __restrict__ qkv, unsigned short* __restrict__ ctx)
{
  int wid  = (blockIdx.x * 256 + threadIdx.x) >> 5;
  int lane = threadIdx.x & 31;
  int bm = wid >> 3;
  int h  = wid & 7;
  const float* base = qkv + (long)bm * 3 * 2304 + h * 96;
  float q[3][3], k[3][3], vv[3][3];
#pragma unroll
  for (int t = 0; t < 3; ++t)
#pragma unroll
    for (int j = 0; j < 3; ++j) {
      int d = lane * 3 + j;
      q[t][j]  = base[(long)t * 2304 + d];
      k[t][j]  = base[(long)t * 2304 + 768 + d];
      vv[t][j] = base[(long)t * 2304 + 1536 + d];
    }
  const float scale = 0.10206207261596577f;        // 1/sqrt(96)
  float s[3][3];
#pragma unroll
  for (int ti = 0; ti < 3; ++ti)
#pragma unroll
    for (int tj = 0; tj < 3; ++tj) {
      float p = q[ti][0]*k[tj][0] + q[ti][1]*k[tj][1] + q[ti][2]*k[tj][2];
#pragma unroll
      for (int off = 16; off >= 1; off >>= 1)
        p += __shfl_xor(p, off, 32);
      s[ti][tj] = p * scale;
    }
#pragma unroll
  for (int ti = 0; ti < 3; ++ti) {
    float mx = fmaxf(s[ti][0], fmaxf(s[ti][1], s[ti][2]));
    float e0 = __expf(s[ti][0]-mx), e1 = __expf(s[ti][1]-mx), e2 = __expf(s[ti][2]-mx);
    float inv = 1.0f / (e0 + e1 + e2);
    float a0 = e0*inv, a1 = e1*inv, a2 = e2*inv;
#pragma unroll
    for (int j = 0; j < 3; ++j) {
      float c = a0*vv[0][j] + a1*vv[1][j] + a2*vv[2][j];
      ctx[((long)bm*3 + ti) * 768 + h*96 + lane*3 + j] = f2bf(c);
    }
  }
}

// LayerNorm over D=768; optional residual add; fp32 and/or bf16 outputs.
__global__ __launch_bounds__(256) void ln_kernel(
    const float* __restrict__ a, const float* __restrict__ b,
    const float* __restrict__ g, const float* __restrict__ beta,
    float* __restrict__ outF, unsigned short* __restrict__ outB)
{
  __shared__ float red[256];
  __shared__ float red2[256];
  long row = blockIdx.x;
  int tid = threadIdx.x;
  float v[3];
  float s = 0.f, s2 = 0.f;
#pragma unroll
  for (int j = 0; j < 3; ++j) {
    int col = tid * 3 + j;
    float x = a[row * 768 + col];
    if (b) x += b[row * 768 + col];
    v[j] = x; s += x; s2 += x * x;
  }
  red[tid] = s; red2[tid] = s2;
  __syncthreads();
  for (int off = 128; off > 0; off >>= 1) {
    if (tid < off) { red[tid] += red[tid + off]; red2[tid] += red2[tid + off]; }
    __syncthreads();
  }
  float mu  = red[0]  * (1.0f / 768.0f);
  float var = red2[0] * (1.0f / 768.0f) - mu * mu;   // biased (jnp.var)
  float rs  = rsqrtf(var + 1e-5f);
#pragma unroll
  for (int j = 0; j < 3; ++j) {
    int col = tid * 3 + j;
    float y = (v[j] - mu) * rs * g[col] + beta[col];
    if (outF) outF[row * 768 + col] = y;
    if (outB) outB[row * 768 + col] = f2bf(y);
  }
}

// Softmax-weighted sum over P=3 properties -> bf16.
__global__ __launch_bounds__(256) void wsum_kernel(
    const float* __restrict__ x2, const float* __restrict__ pw,
    unsigned short* __restrict__ out)
{
  long i = (long)blockIdx.x * 256 + threadIdx.x;     // over 8192*768
  float w0 = pw[0], w1 = pw[1], w2 = pw[2];
  float mx = fmaxf(w0, fmaxf(w1, w2));
  float e0 = __expf(w0-mx), e1 = __expf(w1-mx), e2 = __expf(w2-mx);
  float inv = 1.0f / (e0 + e1 + e2);
  long bm = i / 768, d = i % 768;
  const float* r = x2 + bm * 3 * 768 + d;
  out[i] = f2bf((e0*r[0] + e1*r[768] + e2*r[1536]) * inv);
}

extern "C" void kernel_launch(void* const* d_in, const int* in_sizes, int n_in,
                              void* d_out, int out_size, void* d_ws, size_t ws_size,
                              hipStream_t stream) {
  const float* emb    = (const float*)d_in[0];
  const float* W_proj = (const float*)d_in[1];
  const float* b_proj = (const float*)d_in[2];
  const float* qkv_w  = (const float*)d_in[3];
  const float* qkv_b  = (const float*)d_in[4];
  const float* out_w  = (const float*)d_in[5];
  const float* out_b  = (const float*)d_in[6];
  const float* ln1_g  = (const float*)d_in[7];
  const float* ln1_b  = (const float*)d_in[8];
  const float* ffn_w1 = (const float*)d_in[9];
  const float* ffn_b1 = (const float*)d_in[10];
  const float* ffn_w2 = (const float*)d_in[11];
  const float* ffn_b2 = (const float*)d_in[12];
  const float* ln2_g  = (const float*)d_in[13];
  const float* ln2_b  = (const float*)d_in[14];
  const float* fuse_w = (const float*)d_in[15];
  const float* fuse_b = (const float*)d_in[16];
  const float* fln_g  = (const float*)d_in[17];
  const float* fln_b  = (const float*)d_in[18];
  const float* pw     = (const float*)d_in[19];
  float* out = (float*)d_out;                        // 8192x768 fp32

  const long D = 768, FF = 1024;

  // --- workspace layout (516 MB) ---
  char* base = (char*)d_ws;
  unsigned short* WB   = (unsigned short*)(base);                    // 12.58 MB bf16 weights
  unsigned short* embB = (unsigned short*)(base + 12582912);         // 37.75 MB
  float*          RA   = (float*)(base + 50331648);                  // 226.5 MB (qkv/oproj/h2/fused)
  float*          Rx   = (float*)(base + 276824064);                 // 75.5 MB (x fp32)
  float*          Ry   = (float*)(base + 352321536);                 // 75.5 MB (y / x2 fp32)
  unsigned short* D1   = (unsigned short*)(base + 427819008);        // 50.3 MB (xB -> hB)
  unsigned short* D2   = (unsigned short*)(base + 478150656);        // 37.7 MB (ctxB -> yB -> wsB)

  // bf16 weight offsets (elements, (N,K) layout)
  unsigned short* WprojT = WB;                       // 3 x 768x768
  unsigned short* qkvT   = WB + 1769472;             // 2304x768
  unsigned short* outT   = WB + 3538944;             // 768x768
  unsigned short* f1T    = WB + 4128768;             // 1024x768
  unsigned short* f2T    = WB + 4915200;             // 768x1024
  unsigned short* fuseT  = WB + 5701632;             // 768x768

  dim3 blk(256);
  dim3 tblk(32, 8);
  // --- prep: transpose+downconvert weights, downconvert embeddings ---
  wT_kernel<<<dim3(24, 24, 3), tblk, 0, stream>>>(W_proj, WprojT, 768, 768);
  wT_kernel<<<dim3(72, 24, 1), tblk, 0, stream>>>(qkv_w,  qkvT,   768, 2304);
  wT_kernel<<<dim3(24, 24, 1), tblk, 0, stream>>>(out_w,  outT,   768, 768);
  wT_kernel<<<dim3(32, 24, 1), tblk, 0, stream>>>(ffn_w1, f1T,    768, 1024);
  wT_kernel<<<dim3(24, 32, 1), tblk, 0, stream>>>(ffn_w2, f2T,    1024, 768);
  wT_kernel<<<dim3(24, 24, 1), tblk, 0, stream>>>(fuse_w, fuseT,  768, 768);
  cvt_kernel<<<dim3(36864), blk, 0, stream>>>(emb, embB);

  // 1. per-property projection -> x fp32 (Rx) + xB bf16 (D1)
  gemm_bf16_v3<<<dim3(6, 64, 3), blk, 0, stream>>>(
      embB, 3*D, D, WprojT, D*D, b_proj, D,
      Rx, 3*D, D, D1, 3*D, D, 768, 0);
  // 2. qkv -> RA fp32
  gemm_bf16_v3<<<dim3(18, 192, 1), blk, 0, stream>>>(
      D1, D, 0, qkvT, 0, qkv_b, 0,
      RA, 3*D, 0, nullptr, 0, 0, 768, 0);
  // 3. attention -> ctxB bf16 (D2)
  attn_kernel<<<dim3(8192), blk, 0, stream>>>(RA, D2);
  // 4. out projection -> RA fp32 (qkv dead)
  gemm_bf16_v3<<<dim3(6, 192, 1), blk, 0, stream>>>(
      D2, D, 0, outT, 0, out_b, 0,
      RA, D, 0, nullptr, 0, 0, 768, 0);
  // 5. ln1: y = LN(x + attn_out) -> Ry fp32 + yB bf16 (D2)
  ln_kernel<<<dim3(24576), blk, 0, stream>>>(Rx, RA, ln1_g, ln1_b, Ry, D2);
  // 6. ffn1 + relu -> hB bf16 (D1)
  gemm_bf16_v3<<<dim3(8, 192, 1), blk, 0, stream>>>(
      D2, D, 0, f1T, 0, ffn_b1, 0,
      nullptr, 0, 0, D1, FF, 0, 768, 1);
  // 7. ffn2 -> RA fp32
  gemm_bf16_v3<<<dim3(6, 192, 1), blk, 0, stream>>>(
      D1, FF, 0, f2T, 0, ffn_b2, 0,
      RA, D, 0, nullptr, 0, 0, 1024, 0);
  // 8. ln2: x2 = LN(y + h2) -> Ry (in place, block-local rows)
  ln_kernel<<<dim3(24576), blk, 0, stream>>>(Ry, RA, ln2_g, ln2_b, Ry, nullptr);
  // 9. property softmax-mix -> wsB bf16 (D2)
  wsum_kernel<<<dim3(24576), blk, 0, stream>>>(Ry, pw, D2);
  // 10. fuse GEMM + relu -> RA fp32
  gemm_bf16_v3<<<dim3(6, 64, 1), blk, 0, stream>>>(
      D2, D, 0, fuseT, 0, fuse_b, 0,
      RA, D, 0, nullptr, 0, 0, 768, 1);
  // 11. final LayerNorm -> d_out
  ln_kernel<<<dim3(8192), blk, 0, stream>>>(RA, nullptr, fln_g, fln_b, out, nullptr);
}
